// MemoryEfficientAttention_27281632264461
// MI455X (gfx1250) — matrix operand order repaired and checked by reading, not compiled, not run on hardware
//
#include <hip/hip_runtime.h>
#include <hip/hip_bf16.h>

// Problem constants (from reference)
#define BB 2
#define SS 4096
#define DD 1024
#define HH 16
#define HD 64

typedef __attribute__((ext_vector_type(16))) _Float16 v16h;
typedef __attribute__((ext_vector_type(8)))  _Float16 v8h;
typedef __attribute__((ext_vector_type(4)))  _Float16 v4h;
typedef __attribute__((ext_vector_type(8)))  float    v8f;
typedef __attribute__((ext_vector_type(4)))  int      v4i;

static __device__ __forceinline__ v8f zero8() {
  v8f z = {0.f, 0.f, 0.f, 0.f, 0.f, 0.f, 0.f, 0.f};
  return z;
}

// ---------------------------------------------------------------------------
// Async global->LDS copy (CDNA5 GLOBAL_LOAD_ASYNC_TO_LDS_B128, ASYNCcnt).
// Probe-confirmed signature: param0 = v4i* in AS1 (__device__),
// param1 = v4i* in AS3 (__shared__), then two int (offset, cpol) args.
// Guarded so the file still compiles if the toolchain doesn't expose it.
// ---------------------------------------------------------------------------
#if __has_builtin(__builtin_amdgcn_global_load_async_to_lds_b128)
#define HAVE_ASYNC 1
#else
#define HAVE_ASYNC 0
#endif

static __device__ __forceinline__ void g2l_b128(const _Float16* g, _Float16* l) {
#if HAVE_ASYNC
  // generic->AS1 and generic->AS3 via integer casts (low 32 bits of a generic
  // LDS pointer are the LDS byte address on gfx1250).
  __builtin_amdgcn_global_load_async_to_lds_b128(
      (__attribute__((address_space(1))) v4i*)(unsigned long long)(g),
      (__attribute__((address_space(3))) v4i*)(unsigned int)(unsigned long long)(l),
      0, 0);
#else
  *(v8h*)l = *(const v8h*)g;
#endif
}

static __device__ __forceinline__ void wait_async_all() {
#if HAVE_ASYNC
#if __has_builtin(__builtin_amdgcn_s_wait_asynccnt)
  __builtin_amdgcn_s_wait_asynccnt(0);
#else
  asm volatile("s_wait_asynccnt 0x0" ::: "memory");
#endif
#endif
}

// ---------------------------------------------------------------------------
// fp32 -> f16 conversion, 4-wide
// ---------------------------------------------------------------------------
__global__ __launch_bounds__(256) void f32_to_f16_kernel(
    const float* __restrict__ in, _Float16* __restrict__ out, int n4) {
  int i = blockIdx.x * blockDim.x + threadIdx.x;
  if (i >= n4) return;
  float4 f = ((const float4*)in)[i];
  v4h h;
  h[0] = (_Float16)f.x; h[1] = (_Float16)f.y;
  h[2] = (_Float16)f.z; h[3] = (_Float16)f.w;
  ((v4h*)out)[i] = h;
}

// ---------------------------------------------------------------------------
// GEMM: C[M,N] = A[M,K] @ W[N,K]^T + bias   (torch Linear)
// MODE 0: f16 row-major out [M,N]
// MODE 1: fp32 row-major out [M,N]
// MODE 2: f16 out transposed per head: [b][h][hd][s]  (for V)
// Block 256 thr = 8 waves; block tile 64(M) x 128(N); wave tile 32x32.
// Working set is L2-resident (192 MB L2) -> fragments load straight from
// global; each fragment reused across the 2x2 tile.
// ---------------------------------------------------------------------------
template <int MODE>
__global__ __launch_bounds__(256) void gemm_wmma_kernel(
    const _Float16* __restrict__ A, const _Float16* __restrict__ W,
    const float* __restrict__ bias, void* __restrict__ Cout,
    int M, int N, int K) {
  const int lane = threadIdx.x & 31;
  const int wave = threadIdx.x >> 5;
  const int wm = wave & 1;   // 2 wave rows
  const int wn = wave >> 1;  // 4 wave cols
  const int l16 = lane & 15;
  const int lh  = lane >> 4; // half-wave select

  const int blockM = blockIdx.y * 64;
  const int blockN = blockIdx.x * 128;
  const int rowA0 = blockM + wm * 32 + l16;
  const int colB0 = blockN + wn * 32 + l16;

  v8f acc[2][2];
#pragma unroll
  for (int i = 0; i < 2; i++)
#pragma unroll
    for (int j = 0; j < 2; j++) acc[i][j] = zero8();

  for (int kc = 0; kc < K; kc += 32) {
    v16h a[2], b[2];
#pragma unroll
    for (int mt = 0; mt < 2; mt++) {
      // A fragment 16x32: lane l16 = row m; elems i -> K = 16*lh + i (contig)
      const _Float16* p = A + (size_t)(rowA0 + mt * 16) * K + kc;
      v8h lo = *(const v8h*)(p + 16 * lh);
      v8h hi = *(const v8h*)(p + 16 * lh + 8);
#pragma unroll
      for (int i = 0; i < 8; i++) { a[mt][i] = lo[i]; a[mt][i + 8] = hi[i]; }
    }
#pragma unroll
    for (int nt = 0; nt < 2; nt++) {
      // B fragment 32x16: lane l16 = col n; elems i -> K = 16*lh + i (contig)
      const _Float16* p = W + (size_t)(colB0 + nt * 16) * K + kc + 16 * lh;
      b[nt] = *(const v16h*)p;
    }
#pragma unroll
    for (int mt = 0; mt < 2; mt++)
#pragma unroll
      for (int nt = 0; nt < 2; nt++)
        acc[mt][nt] = __builtin_amdgcn_wmma_f32_16x16x32_f16(
            false, a[mt], false, b[nt], (short)0, acc[mt][nt], false, false);
  }

#pragma unroll
  for (int mt = 0; mt < 2; mt++) {
#pragma unroll
    for (int nt = 0; nt < 2; nt++) {
      const int n = colB0 + nt * 16;
      const float bn = bias[n];
#pragma unroll
      for (int r = 0; r < 8; r++) {
        const int m = blockM + wm * 32 + mt * 16 + r + 8 * lh;
        const float v = acc[mt][nt][r] + bn;
        if (MODE == 0) {
          ((_Float16*)Cout)[(size_t)m * N + n] = (_Float16)v;
        } else if (MODE == 1) {
          ((float*)Cout)[(size_t)m * N + n] = v;
        } else {
          // V transposed: [b][h][hd][s], s contiguous
          const int b_ = m >> 12, s_ = m & (SS - 1);
          const int h_ = n >> 6, hd_ = n & (HD - 1);
          ((_Float16*)Cout)[(((size_t)(b_ * HH + h_) * HD + hd_) * SS) + s_] =
              (_Float16)v;
        }
      }
    }
  }
}

// ---------------------------------------------------------------------------
// Flash attention: each wave owns one 16-row Q tile; 8 waves per block share
// one (b,h). K chunk [key][hd] and pre-transposed V chunk [hd][key] are DMA'd
// (async global->LDS when available) into double-buffered LDS so the next
// chunk's copy overlaps this chunk's 18 WMMAs. Online softmax in fp32; the
// row-sum of P is computed on the matrix pipe as P @ ones.
// ---------------------------------------------------------------------------
__global__ __launch_bounds__(256) void attn_kernel(
    const _Float16* __restrict__ Qh, const _Float16* __restrict__ Kh,
    const _Float16* __restrict__ Vt, _Float16* __restrict__ Ah) {
  __shared__ _Float16 Kt[2][64 * 64];    // [key][hd]  2 x 8 KB
  __shared__ _Float16 Vl[2][64 * 64];    // [hd][key]  2 x 8 KB
  __shared__ _Float16 Pbuf[8][16 * 64];  // per-wave P 16 KB

  const int tid  = threadIdx.x;
  const int lane = tid & 31;
  const int wave = tid >> 5;
  const int l16  = lane & 15;
  const int lh   = lane >> 4;

  const int bh = blockIdx.x >> 5;  // 32 q-groups per (b,h)
  const int qg = blockIdx.x & 31;
  const int b  = bh >> 4;          // H = 16
  const int h  = bh & 15;

  const int qbase = (qg * 8 + wave) * 16;
  const size_t headoff = (size_t)b * SS * DD + (size_t)h * HD;  // Q/K/A layout
  const _Float16* Kg = Kh + headoff;
  const _Float16* Vg = Vt + (size_t)(b * HH + h) * HD * SS;     // [hd][s]

  // tile-copy assignments: 512 b128 segments per 64x64 tile, 2 per thread
  const int seg0 = tid, seg1 = tid + 256;

  // Load the wave's Q tile as two A fragments (hd 0..31, 32..63); kept live.
  v16h aq[2];
  {
    const _Float16* qp = Qh + headoff + (size_t)(qbase + l16) * DD;
#pragma unroll
    for (int c = 0; c < 2; c++) {
      v8h lo = *(const v8h*)(qp + c * 32 + 16 * lh);
      v8h hi = *(const v8h*)(qp + c * 32 + 16 * lh + 8);
#pragma unroll
      for (int i = 0; i < 8; i++) { aq[c][i] = lo[i]; aq[c][i + 8] = hi[i]; }
    }
  }

  v16h ones;
#pragma unroll
  for (int i = 0; i < 16; i++) ones[i] = (_Float16)1.0f;

  float rm[8], rl[8];
  v8f o[4];
#pragma unroll
  for (int r = 0; r < 8; r++) { rm[r] = -__builtin_inff(); rl[r] = 0.f; }
#pragma unroll
  for (int nt = 0; nt < 4; nt++) o[nt] = zero8();

  const float scale = 0.125f;  // 1/sqrt(HD)

  // ---- prologue: stage chunk 0 into buffer 0 ----
  {
#pragma unroll
    for (int ss = 0; ss < 2; ss++) {
      const int seg = ss ? seg1 : seg0;
      const int row = seg >> 3;            // 0..63
      const int c8  = (seg & 7) * 8;       // halves offset
      g2l_b128(Kg + (size_t)row * DD + c8, &Kt[0][row * 64 + c8]);
      g2l_b128(Vg + (size_t)row * SS + c8, &Vl[0][row * 64 + c8]);
    }
  }
  wait_async_all();
  __syncthreads();

  int cur = 0;
  for (int kc = 0; kc < SS; kc += 64) {
    // ---- prefetch next chunk into the other buffer ----
    if (kc + 64 < SS) {
      const int nb = cur ^ 1;
#pragma unroll
      for (int ss = 0; ss < 2; ss++) {
        const int seg = ss ? seg1 : seg0;
        const int row = seg >> 3;
        const int c8  = (seg & 7) * 8;
        g2l_b128(Kg + (size_t)(kc + 64 + row) * DD + c8, &Kt[nb][row * 64 + c8]);
        g2l_b128(Vg + (size_t)row * SS + (kc + 64) + c8, &Vl[nb][row * 64 + c8]);
      }
    }

    // ---- scores S = Q K^T : 4 fragments of 16 keys, 2 WMMAs each ----
    v8f sc[4];
#pragma unroll
    for (int kt = 0; kt < 4; kt++) {
      v16h bk0 = *(const v16h*)(&Kt[cur][(kt * 16 + l16) * 64 + 16 * lh]);
      v16h bk1 = *(const v16h*)(&Kt[cur][(kt * 16 + l16) * 64 + 32 + 16 * lh]);
      v8f s = zero8();
      s = __builtin_amdgcn_wmma_f32_16x16x32_f16(false, aq[0], false, bk0,
                                                 (short)0, s, false, false);
      s = __builtin_amdgcn_wmma_f32_16x16x32_f16(false, aq[1], false, bk1,
                                                 (short)0, s, false, false);
      sc[kt] = s;
    }

    // ---- online softmax max: row stats replicated across each half-wave,
    //      register r <-> row (r + 8*lh), matching the C/D fragment layout ----
    float corr[8];
#pragma unroll
    for (int r = 0; r < 8; r++) {
      float mx = fmaxf(fmaxf(sc[0][r], sc[1][r]), fmaxf(sc[2][r], sc[3][r]));
      mx *= scale;
#pragma unroll
      for (int m = 1; m < 16; m <<= 1) mx = fmaxf(mx, __shfl_xor(mx, m, 32));
      const float nm = fmaxf(rm[r], mx);
      corr[r] = __expf(rm[r] - nm);
      rm[r] = nm;
    }

    // ---- P = exp(scale*S - rm), written to per-wave LDS ----
    _Float16* Pw = &Pbuf[wave][0];
#pragma unroll
    for (int kt = 0; kt < 4; kt++) {
#pragma unroll
      for (int r = 0; r < 8; r++) {
        const float p = __expf(sc[kt][r] * scale - rm[r]);
        Pw[(r + 8 * lh) * 64 + kt * 16 + l16] = (_Float16)p;
      }
    }
#pragma unroll
    for (int nt = 0; nt < 4; nt++)
#pragma unroll
      for (int r = 0; r < 8; r++) o[nt][r] *= corr[r];

    // ---- O += P @ V, and row-sum(P) = P @ ones on the matrix pipe ----
    v8f lsum = zero8();
#pragma unroll
    for (int c2 = 0; c2 < 2; c2++) {
      v16h ap;
      {
        const _Float16* pp = Pw + (size_t)l16 * 64 + c2 * 32;
        v8h lo = *(const v8h*)(pp + 16 * lh);
        v8h hi = *(const v8h*)(pp + 16 * lh + 8);
#pragma unroll
        for (int i = 0; i < 8; i++) { ap[i] = lo[i]; ap[i + 8] = hi[i]; }
      }
      lsum = __builtin_amdgcn_wmma_f32_16x16x32_f16(false, ap, false, ones,
                                                    (short)0, lsum, false, false);
#pragma unroll
      for (int nt = 0; nt < 4; nt++) {
        v16h bv = *(const v16h*)(&Vl[cur][(nt * 16 + l16) * 64 + c2 * 32 + 16 * lh]);
        o[nt] = __builtin_amdgcn_wmma_f32_16x16x32_f16(false, ap, false, bv,
                                                       (short)0, o[nt], false, false);
      }
    }
#pragma unroll
    for (int r = 0; r < 8; r++) rl[r] = rl[r] * corr[r] + lsum[r];

    // next buffer must be fully landed; all waves done with `cur`
    wait_async_all();
    __syncthreads();
    cur ^= 1;
  }

  // ---- normalize and write attention output (f16, [B,S,D]) ----
  float inv[8];
#pragma unroll
  for (int r = 0; r < 8; r++) inv[r] = 1.0f / rl[r];
#pragma unroll
  for (int nt = 0; nt < 4; nt++) {
#pragma unroll
    for (int r = 0; r < 8; r++) {
      const int m = qbase + r + 8 * lh;
      const int col = nt * 16 + l16;
      Ah[headoff + (size_t)m * DD + col] = (_Float16)(o[nt][r] * inv[r]);
    }
  }
}

// ---------------------------------------------------------------------------
// Launch
// ---------------------------------------------------------------------------
extern "C" void kernel_launch(void* const* d_in, const int* in_sizes, int n_in,
                              void* d_out, int out_size, void* d_ws, size_t ws_size,
                              hipStream_t stream) {
  const float* x  = (const float*)d_in[0];
  const float* Wq = (const float*)d_in[1];
  const float* bq = (const float*)d_in[2];
  const float* Wk = (const float*)d_in[3];
  const float* bk = (const float*)d_in[4];
  const float* Wv = (const float*)d_in[5];
  const float* bv = (const float*)d_in[6];
  const float* Wo = (const float*)d_in[7];
  const float* bo = (const float*)d_in[8];

  const size_t MD = (size_t)BB * SS * DD;  // 8388608
  const size_t WD = (size_t)DD * DD;       // 1048576

  // Workspace layout (f16): xh, 4 weights, Q, K, V^T, attn-out => ~92.3 MB
  _Float16* ws  = (_Float16*)d_ws;
  _Float16* xh  = ws; ws += MD;
  _Float16* wqh = ws; ws += WD;
  _Float16* wkh = ws; ws += WD;
  _Float16* wvh = ws; ws += WD;
  _Float16* woh = ws; ws += WD;
  _Float16* qh  = ws; ws += MD;
  _Float16* kh  = ws; ws += MD;
  _Float16* vt  = ws; ws += MD;  // transposed: [b][h][hd][s]
  _Float16* ah  = ws; ws += MD;

  const int M = BB * SS;  // 8192

  // fp32 -> f16 conversions
  {
    int n4 = (int)(MD / 4);
    f32_to_f16_kernel<<<dim3((n4 + 255) / 256), dim3(256), 0, stream>>>(x, xh, n4);
    n4 = (int)(WD / 4);
    dim3 g((n4 + 255) / 256);
    f32_to_f16_kernel<<<g, dim3(256), 0, stream>>>(Wq, wqh, n4);
    f32_to_f16_kernel<<<g, dim3(256), 0, stream>>>(Wk, wkh, n4);
    f32_to_f16_kernel<<<g, dim3(256), 0, stream>>>(Wv, wvh, n4);
    f32_to_f16_kernel<<<g, dim3(256), 0, stream>>>(Wo, woh, n4);
  }

  // Q/K/V projections (V written pre-transposed per head)
  dim3 gg(DD / 128, M / 64);  // (8, 128)
  gemm_wmma_kernel<0><<<gg, dim3(256), 0, stream>>>(xh, wqh, bq, qh, M, DD, DD);
  gemm_wmma_kernel<0><<<gg, dim3(256), 0, stream>>>(xh, wkh, bk, kh, M, DD, DD);
  gemm_wmma_kernel<2><<<gg, dim3(256), 0, stream>>>(xh, wvh, bv, vt, M, DD, DD);

  // attention: B*H*(S/16/8) blocks of 256 threads
  attn_kernel<<<dim3(BB * HH * 32), dim3(256), 0, stream>>>(qh, kh, vt, ah);

  // output projection (fp32 output + bias)
  gemm_wmma_kernel<1><<<gg, dim3(256), 0, stream>>>(ah, woh, bo, d_out, M, DD, DD);
}
